// PathDependentGenerator_62371515072579
// MI455X (gfx1250) — matrix operand order repaired and checked by reading, not compiled, not run on hardware
//
#include <hip/hip_runtime.h>
#include <hip/hip_bf16.h>
#include <stdint.h>

// ---------------------------------------------------------------------------
// Problem constants: H=64, P=4, N=16, D=8, T=64, MLPW=512, B=2048,
// D_IN = 321 -> padded to 384 (multiple of 64 for BK=64 WMMA K-tiling).
// Drift + diffusion MLPs fused into block-diagonal combined GEMMs:
//   L1: tx(2048x384)        @ [fW0|gW0] -> h (2048x1024)   relu f16
//   L2: h  (block-diag f/g) @ [fW1|gW1] -> h'(2048x1024)   relu f16
//   L3: h' (block-diag f/g) @ [fW2|gW2] -> h (2048x1024)   relu f16
//   L4: h  (block-diag f/g) @ [fW3|gW3] -> out(2048x1088)  tanh f32
// ---------------------------------------------------------------------------
#define BATCH   2048
#define HDIM    64
#define PCOEF   4
#define NNOISE  16
#define DOUT    8
#define TSTEPS  64
#define MLPW    512
#define DIN_PAD 384
#define NCOMB   (2 * MLPW)              // 1024
#define NLAST   (HDIM + HDIM * NNOISE)  // 1088

typedef __attribute__((ext_vector_type(16))) _Float16 v16h;
typedef __attribute__((ext_vector_type(8)))  _Float16 v8h;
typedef __attribute__((ext_vector_type(8)))  float    v8f;
typedef __attribute__((ext_vector_type(4)))  int      v4i;

// CDNA5 async global->LDS copy path (ASYNCcnt-tracked DMA), if the toolchain
// exposes the builtins. Falls back to explicit load+ds_store otherwise.
// Builtin signature (probed from clang diagnostics): pointee type is v4i.
#if defined(__gfx1250__) &&                                             \
    __has_builtin(__builtin_amdgcn_global_load_async_to_lds_b128) &&    \
    __has_builtin(__builtin_amdgcn_s_wait_asynccnt)
#define USE_ASYNC_LDS 1
#define ASYNC_CP16(gp, lp)                                              \
    __builtin_amdgcn_global_load_async_to_lds_b128(                     \
        (__attribute__((address_space(1))) v4i*)(unsigned long long)(const void*)(gp), \
        (__attribute__((address_space(3))) v4i*)                        \
            (__attribute__((address_space(3))) _Float16*)(lp),          \
        0, 0)
#else
#define USE_ASYNC_LDS 0
#endif

// ---------------------------------------------------------------------------
// Weight prep: W (K x N, f32, row-major) -> WT (N x Kpad, f16, K-contiguous)
// ---------------------------------------------------------------------------
__global__ void prep_wt_kernel(const float* __restrict__ W, _Float16* __restrict__ WT,
                               int K, int N, int Kpad) {
    int i = blockIdx.x * blockDim.x + threadIdx.x;
    if (i >= N * Kpad) return;
    int n  = i / Kpad;
    int kk = i - n * Kpad;
    WT[i] = (kk < K) ? (_Float16)W[(size_t)kk * N + n] : (_Float16)0.0f;
}

__global__ void init_hist_kernel(float* __restrict__ hist) {
    int i = blockIdx.x * blockDim.x + threadIdx.x;   // over BATCH*HDIM
    hist[i] = 1.0f;                                  // x0 = ones
}

// ---------------------------------------------------------------------------
// Feature kernel: build tx[b][0..383] (f16) for step k.
//   col 0: t (=k); col 1..64: hist[k,b,:]; col 65..320: real ortho-rfft
//   coeffs (p*64+h, zero beyond bin n//2); col 321..383: zero pad.
// ---------------------------------------------------------------------------
__global__ __launch_bounds__(DIN_PAD)
void feature_kernel(const float* __restrict__ hist, _Float16* __restrict__ tx, int k) {
    __shared__ float sh[TSTEPS * HDIM];   // [t][h], 16 KB
    const int b = blockIdx.x;
    const int n = k + 1;

    for (int i = threadIdx.x; i < n * HDIM; i += blockDim.x) {
        int t = i >> 6, h = i & 63;
        sh[i] = hist[(size_t)t * (BATCH * HDIM) + (size_t)b * HDIM + h];
    }
    __syncthreads();

    const int col = threadIdx.x;
    float v = 0.0f;
    if (col == 0) {
        v = (float)k;
    } else if (col <= HDIM) {
        v = sh[k * HDIM + (col - 1)];
    } else if (col <= HDIM + PCOEF * HDIM) {
        int idx = col - (HDIM + 1);
        int p = idx >> 6;
        int h = idx & 63;
        if (2 * p <= n) {                 // valid rfft bin: p <= n//2
            float w = 6.28318530717958647692f * (float)p / (float)n;
            float s = 0.0f;
            for (int t = 0; t < n; ++t)
                s += sh[t * HDIM + h] * __cosf(w * (float)t);
            v = s * rsqrtf((float)n);     // norm="ortho"
        }
    }
    tx[(size_t)b * DIN_PAD + col] = (_Float16)v;
}

// ---------------------------------------------------------------------------
// WMMA GEMM with block-diagonal A selection:
//   C[:, n] = act( A[:, aOff:aOff+Kpad] @ WT[n]^T + bias(n) )
//   aOff = (n >= nSplit) ? aSplitOff : 0; bias from bias0/bias1 split at nSplit.
// Block tile 128x64, 8 waves (4Mx2N), wave tile 32x32 = 2x2 WMMA 16x16, BK=64.
// Double-buffered LDS; async global->LDS DMA pipelined against WMMA compute.
// ---------------------------------------------------------------------------
template <int ACT>   // 0: ReLU -> f16 out; 1: tanh -> f32 out
__global__ __launch_bounds__(256)
void gemm_wmma_kernel(const _Float16* __restrict__ A, int lda,
                      const _Float16* __restrict__ WT,
                      const float* __restrict__ bias0,
                      const float* __restrict__ bias1,
                      int nSplit, int aSplitOff,
                      void* __restrict__ Cout, int N, int Kpad) {
#if USE_ASYNC_LDS
    __shared__ __align__(16) _Float16 As[2][128][64];   // 32 KB
    __shared__ __align__(16) _Float16 Bs[2][64][64];    // 16 KB
#else
    __shared__ __align__(16) _Float16 As[1][128][64];
    __shared__ __align__(16) _Float16 Bs[1][64][64];
#endif

    const int tid  = threadIdx.x;
    const int lane = tid & 31;
    const int wave = tid >> 5;
    const int wm   = wave >> 1;     // 0..3
    const int wn   = wave & 1;      // 0..1
    const int m16  = lane & 15;
    const int hi   = lane >> 4;     // 0/1

    const int blockM = blockIdx.x * 128;
    const int blockN = blockIdx.y * 64;
    const int aOff   = (blockN >= nSplit) ? aSplitOff : 0;

    // A tile: 128x64 halfs = 1024 16B-chunks -> 4 per thread
    const int arow = tid >> 1;               // 0..127
    const int acol = (tid & 1) * 8;          // 0/8 (then +16 per chunk)
    // B tile: 64x64 halfs = 512 chunks -> 2 per thread
    const int brow = tid >> 2;               // 0..63
    const int bcol = (tid & 3) * 8;          // 0,8,16,24 (then +32)

    const _Float16* Abase = A  + (size_t)(blockM + arow) * lda + aOff + acol;
    const _Float16* Bbase = WT + (size_t)(blockN + brow) * Kpad + bcol;

    v8f acc[2][2] = {};
    const int nk = Kpad / 64;

#if USE_ASYNC_LDS
    auto issue_tile = [&](int i, int buf) {
        const _Float16* Ag = Abase + i * 64;
        const _Float16* Bg = Bbase + i * 64;
#pragma unroll
        for (int j = 0; j < 4; ++j)
            ASYNC_CP16(Ag + 16 * j, &As[buf][arow][acol + 16 * j]);
#pragma unroll
        for (int j = 0; j < 2; ++j)
            ASYNC_CP16(Bg + 32 * j, &Bs[buf][brow][bcol + 32 * j]);
    };
    issue_tile(0, 0);
#endif

    for (int i = 0; i < nk; ++i) {
#if USE_ASYNC_LDS
        const int cur = i & 1;
        if (i + 1 < nk) {
            issue_tile(i + 1, cur ^ 1);                 // DMA next tile
            __builtin_amdgcn_s_wait_asynccnt(6);        // current tile done
        } else {
            __builtin_amdgcn_s_wait_asynccnt(0);
        }
        __syncthreads();
#else
        const int cur = 0;
        // Batched staging: distinct temporaries so global loads overlap.
        uint4 ta[4], tb[2];
        {
            const _Float16* Ag = Abase + i * 64;
            const _Float16* Bg = Bbase + i * 64;
#pragma unroll
            for (int j = 0; j < 4; ++j) ta[j] = *(const uint4*)(Ag + 16 * j);
#pragma unroll
            for (int j = 0; j < 2; ++j) tb[j] = *(const uint4*)(Bg + 32 * j);
        }
#pragma unroll
        for (int j = 0; j < 4; ++j) *(uint4*)(&As[0][arow][acol + 16 * j]) = ta[j];
#pragma unroll
        for (int j = 0; j < 2; ++j) *(uint4*)(&Bs[0][brow][bcol + 32 * j]) = tb[j];
        __syncthreads();
#endif

#pragma unroll
        for (int ks = 0; ks < 64; ks += 32) {
            // A fragments (ISA 7.12.2, 16-bit A 16x32):
            // lane<16 -> K {0..7,16..23}; lane>=16 -> K {8..15,24..31}
            v16h af[2];
#pragma unroll
            for (int mt = 0; mt < 2; ++mt) {
                const _Float16* p = &As[cur][wm * 32 + mt * 16 + m16][ks + hi * 8];
                v8h lo = *(const v8h*)p;
                v8h hh = *(const v8h*)(p + 16);
#pragma unroll
                for (int e = 0; e < 8; ++e) { af[mt][e] = lo[e]; af[mt][e + 8] = hh[e]; }
            }
            // B fragments: lane = column n; lane<16 -> K {0..7,16..23}; lane>=16 -> K {8..15,24..31}
            v16h bf[2];
#pragma unroll
            for (int nt = 0; nt < 2; ++nt) {
                const _Float16* p = &Bs[cur][wn * 32 + nt * 16 + m16][ks + hi * 8];
                v8h lo = *(const v8h*)p;
                v8h hh = *(const v8h*)(p + 16);
#pragma unroll
                for (int e = 0; e < 8; ++e) { bf[nt][e] = lo[e]; bf[nt][e + 8] = hh[e]; }
            }
#pragma unroll
            for (int mt = 0; mt < 2; ++mt)
#pragma unroll
                for (int nt = 0; nt < 2; ++nt)
                    acc[mt][nt] = __builtin_amdgcn_wmma_f32_16x16x32_f16(
                        false, af[mt], false, bf[nt], (short)0, acc[mt][nt], false, false);
        }
        __syncthreads();
    }

    // Epilogue: C/D layout -> VGPR r: M = r + 8*hi, N = lane&15.
#pragma unroll
    for (int mt = 0; mt < 2; ++mt) {
#pragma unroll
        for (int nt = 0; nt < 2; ++nt) {
            const int gn = blockN + wn * 32 + nt * 16 + m16;
            const float bv = (gn < nSplit) ? bias0[gn] : bias1[gn - nSplit];
#pragma unroll
            for (int r = 0; r < 8; ++r) {
                const int gm = blockM + wm * 32 + mt * 16 + hi * 8 + r;
                float v = acc[mt][nt][r] + bv;
                if (ACT == 0) {
                    v = fmaxf(v, 0.0f);
                    ((_Float16*)Cout)[(size_t)gm * N + gn] = (_Float16)v;
                } else {
                    ((float*)Cout)[(size_t)gm * N + gn] = tanhf(v);
                }
            }
        }
    }
}

// ---------------------------------------------------------------------------
// Euler-Maruyama update from combined last-layer output (2048 x 1088 f32):
//   cols 0..63 = drift(b,h); cols 64..1087 = diff(b,h,n) at 64 + h*16 + n
// ---------------------------------------------------------------------------
__global__ void update_kernel(float* __restrict__ hist,
                              const float* __restrict__ outc,
                              const float* __restrict__ dbt, int k) {
    int i = blockIdx.x * blockDim.x + threadIdx.x;   // b*64 + h
    int b = i >> 6, h = i & 63;
    const float* row = outc + (size_t)b * NLAST;
    float x = hist[(size_t)k * (BATCH * HDIM) + i] + row[h];
    const float* dB = dbt + (size_t)k * (BATCH * NNOISE) + (size_t)b * NNOISE;
    const float* dF = row + HDIM + h * NNOISE;
#pragma unroll
    for (int nn = 0; nn < NNOISE; ++nn) x += dF[nn] * dB[nn];
    hist[(size_t)(k + 1) * (BATCH * HDIM) + i] = x;
}

// ---------------------------------------------------------------------------
// Readout: out[b,t,0] = ts[t]; out[b,t,1+d] = hist[t,b,:] . rW[:,d] + rb[d]
// ---------------------------------------------------------------------------
__global__ void readout_kernel(const float* __restrict__ hist,
                               const float* __restrict__ rW,
                               const float* __restrict__ rb,
                               const float* __restrict__ ts,
                               float* __restrict__ out) {
    int i = blockIdx.x * blockDim.x + threadIdx.x;   // b*T + t
    int b = i >> 6, t = i & 63;
    const float* xr = hist + (size_t)t * (BATCH * HDIM) + (size_t)b * HDIM;
    float acc[DOUT];
#pragma unroll
    for (int d = 0; d < DOUT; ++d) acc[d] = rb[d];
    for (int h = 0; h < HDIM; ++h) {
        float xv = xr[h];
#pragma unroll
        for (int d = 0; d < DOUT; ++d) acc[d] += xv * rW[h * DOUT + d];
    }
    float* o = out + ((size_t)b * TSTEPS + t) * (1 + DOUT);
    o[0] = ts[t];
#pragma unroll
    for (int d = 0; d < DOUT; ++d) o[1 + d] = acc[d];
}

// ---------------------------------------------------------------------------
// Host launcher
// ---------------------------------------------------------------------------
extern "C" void kernel_launch(void* const* d_in, const int* in_sizes, int n_in,
                              void* d_out, int out_size, void* d_ws, size_t ws_size,
                              hipStream_t stream) {
    (void)in_sizes; (void)n_in; (void)out_size; (void)ws_size;

    const float* fW[4] = {(const float*)d_in[0], (const float*)d_in[2],
                          (const float*)d_in[4], (const float*)d_in[6]};
    const float* fb[4] = {(const float*)d_in[1], (const float*)d_in[3],
                          (const float*)d_in[5], (const float*)d_in[7]};
    const float* gW[4] = {(const float*)d_in[8],  (const float*)d_in[10],
                          (const float*)d_in[12], (const float*)d_in[14]};
    const float* gb[4] = {(const float*)d_in[9],  (const float*)d_in[11],
                          (const float*)d_in[13], (const float*)d_in[15]};
    const float* rW  = (const float*)d_in[16];
    const float* rb  = (const float*)d_in[17];
    const float* ts  = (const float*)d_in[18];
    const float* dbt = (const float*)d_in[19];

    // ---- carve workspace ----
    char* ws = (char*)d_ws;
    size_t off = 0;
    auto take = [&](size_t bytes) -> char* {
        char* p = ws + off;
        off += (bytes + 255) & ~(size_t)255;
        return p;
    };

    _Float16* cWT0 = (_Float16*)take((size_t)NCOMB * DIN_PAD * 2);  // 1024 x 384
    _Float16* cWT1 = (_Float16*)take((size_t)NCOMB * MLPW * 2);     // 1024 x 512
    _Float16* cWT2 = (_Float16*)take((size_t)NCOMB * MLPW * 2);     // 1024 x 512
    _Float16* cWT3 = (_Float16*)take((size_t)NLAST * MLPW * 2);     // 1088 x 512

    _Float16* tx   = (_Float16*)take((size_t)BATCH * DIN_PAD * 2);
    _Float16* hA   = (_Float16*)take((size_t)BATCH * NCOMB * 2);
    _Float16* hB   = (_Float16*)take((size_t)BATCH * NCOMB * 2);
    float*    outc = (float*)take((size_t)BATCH * NLAST * 4);
    float*    hist = (float*)take((size_t)TSTEPS * BATCH * HDIM * 4);

    // ---- weight prep: pack f then g halves into combined buffers ----
    auto prep = [&](const float* W, _Float16* WT, int K, int N, int Kpad) {
        int tot = N * Kpad;
        prep_wt_kernel<<<(tot + 255) / 256, 256, 0, stream>>>(W, WT, K, N, Kpad);
    };
    prep(fW[0], cWT0, 321, MLPW, DIN_PAD);
    prep(gW[0], cWT0 + (size_t)MLPW * DIN_PAD, 321, MLPW, DIN_PAD);
    prep(fW[1], cWT1, MLPW, MLPW, MLPW);
    prep(gW[1], cWT1 + (size_t)MLPW * MLPW, MLPW, MLPW, MLPW);
    prep(fW[2], cWT2, MLPW, MLPW, MLPW);
    prep(gW[2], cWT2 + (size_t)MLPW * MLPW, MLPW, MLPW, MLPW);
    prep(fW[3], cWT3, MLPW, HDIM, MLPW);
    prep(gW[3], cWT3 + (size_t)HDIM * MLPW, MLPW, HDIM * NNOISE, MLPW);

    init_hist_kernel<<<(BATCH * HDIM) / 256, 256, 0, stream>>>(hist);

    // ---- sequential SDE integration: 63 fused steps ----
    const dim3 gL13(BATCH / 128, NCOMB / 64);   // (16,16)
    const dim3 gL4 (BATCH / 128, NLAST / 64);   // (16,17)

    for (int k = 0; k < TSTEPS - 1; ++k) {
        feature_kernel<<<BATCH, DIN_PAD, 0, stream>>>(hist, tx, k);

        gemm_wmma_kernel<0><<<gL13, 256, 0, stream>>>(
            tx, DIN_PAD, cWT0, fb[0], gb[0], MLPW, 0, hA, NCOMB, DIN_PAD);
        gemm_wmma_kernel<0><<<gL13, 256, 0, stream>>>(
            hA, NCOMB, cWT1, fb[1], gb[1], MLPW, MLPW, hB, NCOMB, MLPW);
        gemm_wmma_kernel<0><<<gL13, 256, 0, stream>>>(
            hB, NCOMB, cWT2, fb[2], gb[2], MLPW, MLPW, hA, NCOMB, MLPW);
        gemm_wmma_kernel<1><<<gL4, 256, 0, stream>>>(
            hA, NCOMB, cWT3, fb[3], gb[3], HDIM, MLPW, outc, NLAST, MLPW);

        update_kernel<<<(BATCH * HDIM) / 256, 256, 0, stream>>>(hist, outc, dbt, k);
    }

    readout_kernel<<<(BATCH * TSTEPS) / 256, 256, 0, stream>>>(hist, rW, rb, ts, (float*)d_out);
}
